// Attention_5669356836241
// MI455X (gfx1250) — matrix operand order repaired and checked by reading, not compiled, not run on hardware
//
#include <hip/hip_runtime.h>
#include <hip/hip_bf16.h>

// ---------------------------------------------------------------------------
// Attention forward for MI455X (gfx1250): bf16 WMMA everywhere, flash-style
// fused softmax, async global->LDS staging (ASYNCcnt) with double buffering.
// ---------------------------------------------------------------------------

typedef __bf16 bf16;
typedef __attribute__((ext_vector_type(16))) __bf16 bf16x16;
typedef __attribute__((ext_vector_type(8)))  __bf16 bf16x8;
typedef __attribute__((ext_vector_type(8)))  float  f32x8;
typedef __attribute__((ext_vector_type(4)))  int    int4v;
typedef __attribute__((address_space(1))) int4v* gbl_int4p;   // global int4*
typedef __attribute__((address_space(3))) int4v* lds_int4p;   // LDS int4*

#define BATCH    2
#define SEQ      4096
#define DIM      512
#define HEADS    8
#define HEAD_DIM 64
#define SCALE    0.125f   // 64^-0.5

// ---- CDNA5 async global->LDS path (guarded; falls back to sync copies) ----
#if defined(__has_builtin)
#if __has_builtin(__builtin_amdgcn_global_load_async_to_lds_b128) && \
    __has_builtin(__builtin_amdgcn_s_wait_asynccnt)
#define USE_ASYNC_LDS 1
#endif
#endif
#ifndef USE_ASYNC_LDS
#define USE_ASYNC_LDS 0
#endif

// Copy 16 bytes (8 bf16) global -> LDS, per lane.
__device__ __forceinline__ void cp_g2l_b128(const bf16* g, bf16* l) {
#if USE_ASYNC_LDS
    __builtin_amdgcn_global_load_async_to_lds_b128(
        (gbl_int4p)(bf16*)g, (lds_int4p)l, 0, 0);
#else
    *(bf16x8*)l = *(const bf16x8*)g;
#endif
}

__device__ __forceinline__ void wait_async_all() {
#if USE_ASYNC_LDS
    __builtin_amdgcn_s_wait_asynccnt(0);
#endif
}

__device__ __forceinline__ bf16x16 combine8(bf16x8 lo, bf16x8 hi) {
    return __builtin_shufflevector(lo, hi, 0, 1, 2, 3, 4, 5, 6, 7,
                                           8, 9, 10, 11, 12, 13, 14, 15);
}

// A-fragment (16x32 bf16, MxK): lane = M (lane&15); lanes 0-15 hold K
// {0..15}, lanes 16-31 hold K {16..31}  (contiguous, matches B fragments)
__device__ __forceinline__ bf16x16 load_a_frag(const bf16* rowp, int k0, int half) {
    const bf16* p = rowp + k0 + half * 16;
    bf16x8 lo = *(const bf16x8*)(p);
    bf16x8 hi = *(const bf16x8*)(p + 8);
    return combine8(lo, hi);
}

__device__ __forceinline__ f32x8 wmma_bf16(bf16x16 a, bf16x16 b, f32x8 c) {
    return __builtin_amdgcn_wmma_f32_16x16x32_bf16(
        /*neg_a=*/false, a, /*neg_b=*/false, b,
        /*c_mod=*/(short)0, c, /*reuse_a=*/false, /*reuse_b=*/false);
}

// ---------------------------------------------------------------------------
// f32 -> bf16 conversion, 8 elements per thread (element counts are /8)
// ---------------------------------------------------------------------------
__global__ void f32_to_bf16_kernel(const float* __restrict__ src,
                                   bf16* __restrict__ dst, int n) {
    int i = (blockIdx.x * blockDim.x + threadIdx.x) * 8;
    if (i < n) {
        bf16x8 v;
#pragma unroll
        for (int u = 0; u < 8; u++) v[u] = (bf16)src[i + u];
        *(bf16x8*)(dst + i) = v;
    }
}

// ---------------------------------------------------------------------------
// QKV projection: [8192,512]bf16 @ W^T[512,1536]bf16 + b, routed into
//   Q (pre-scaled by SCALE) : [B,H,N,64]   bf16
//   K                       : [B,H,N,64]   bf16
//   V (pre-transposed!)     : [B,H,64,N]   bf16   -> flash stages V with
//                                                     contiguous copies only
// Block: 256 thr = 8 waves; tile 128(M) x 64(N); weight tile staged in LDS
// in two 64x256 halves (32KB) via async copies.
// ---------------------------------------------------------------------------
__global__ __launch_bounds__(256)
void qkv_gemm_kernel(const bf16* __restrict__ A, const bf16* __restrict__ W,
                     const float* __restrict__ bias,
                     bf16* __restrict__ q, bf16* __restrict__ k,
                     bf16* __restrict__ v) {
    __shared__ bf16 wS[64 * 256];   // 32 KB weight tile half

    const int m0 = blockIdx.x * 128;
    const int n0 = blockIdx.y * 64;
    const int tid = threadIdx.x, wave = tid >> 5, lane = tid & 31;
    const int half = lane >> 4, l15 = lane & 15;
    const int mrow = m0 + wave * 16 + l15;

    const int lrow = tid >> 2;      // 0..63 : weight row staged by this thread
    const int lseg = tid & 3;       // 64-element segment within 256-wide half

    const f32x8 zero = {0.f, 0.f, 0.f, 0.f, 0.f, 0.f, 0.f, 0.f};
    f32x8 acc[4] = {zero, zero, zero, zero};

    for (int kk = 0; kk < DIM; kk += 256) {
        // Stage 64x256 weight half: 8 x b128 per thread, async
#pragma unroll
        for (int u = 0; u < 8; u++)
            cp_g2l_b128(W + (size_t)(n0 + lrow) * DIM + kk + lseg * 64 + u * 8,
                        &wS[lrow * 256 + lseg * 64 + u * 8]);
        wait_async_all();
        __syncthreads();

        for (int k0 = 0; k0 < 256; k0 += 32) {
            bf16x16 afrag = load_a_frag(A + (size_t)mrow * DIM, kk + k0, half);
#pragma unroll
            for (int nt = 0; nt < 4; nt++) {
                bf16x16 bfrag =
                    *(const bf16x16*)&wS[(nt * 16 + l15) * 256 + k0 + half * 16];
                acc[nt] = wmma_bf16(afrag, bfrag, acc[nt]);
            }
        }
        __syncthreads();   // done reading this half before restaging
    }

    const int section = n0 >> 9;            // 0=Q 1=K 2=V (tile never straddles)
    const int h = (n0 & 511) >> 6;          // head index (uniform per block)
    const float scl = section == 0 ? SCALE : 1.0f;

#pragma unroll
    for (int nt = 0; nt < 4; nt++) {
        const int col = n0 + nt * 16 + l15;
        const float bs = bias[col];
        const int d = nt * 16 + l15;        // dim within head
#pragma unroll
        for (int r = 0; r < 8; r++) {       // C-tile: M = r + 8*half
            const int m = m0 + wave * 16 + r + 8 * half;
            const int b = m >> 12, n = m & (SEQ - 1);
            const float val = (acc[nt][r] + bs) * scl;
            if (section < 2) {              // Q/K: [B,H,N,Dh]
                bf16* dst = section == 0 ? q : k;
                dst[(((size_t)(b * HEADS + h)) * SEQ + n) * HEAD_DIM + d] = (bf16)val;
            } else {                        // V: [B,H,Dh,N] (transposed)
                v[(((size_t)(b * HEADS + h)) * HEAD_DIM + d) * SEQ + n] = (bf16)val;
            }
        }
    }
}

// ---------------------------------------------------------------------------
// Flash attention: per (b,h) 128-query tile per block, 64-key column tiles.
// K [key][dim] and V [dim][key] tiles double-buffered in LDS via async
// copies issued one tile ahead. Online softmax state in VGPRs.
// ---------------------------------------------------------------------------
__global__ __launch_bounds__(256)
void flash_attn_kernel(const bf16* __restrict__ Q, const bf16* __restrict__ K,
                       const bf16* __restrict__ V, bf16* __restrict__ O) {
    __shared__ bf16 kT[2][64 * 64];     // [key][dim]
    __shared__ bf16 vT[2][64 * 64];     // [dim][key] (global V already transposed)
    __shared__ bf16 pS[8 * 16 * 64];    // per-wave P re-layout slab

    const int bh = blockIdx.y;                      // 0..15  (b*8+h)
    const int tile = blockIdx.x;                    // 0..31
    const int tid = threadIdx.x, wave = tid >> 5, lane = tid & 31;
    const int half = lane >> 4, l15 = lane & 15;

    const bf16* qb = Q + (size_t)bh * SEQ * HEAD_DIM;
    const bf16* kb = K + (size_t)bh * SEQ * HEAD_DIM;
    const bf16* vb = V + (size_t)bh * HEAD_DIM * SEQ;   // [64][4096]

    // Q fragments for this wave's 16 rows live in VGPRs for whole kernel.
    const int qrow = tile * 128 + wave * 16 + l15;
    bf16x16 qfrag[2];
#pragma unroll
    for (int ks = 0; ks < 2; ks++)
        qfrag[ks] = load_a_frag(qb + (size_t)qrow * HEAD_DIM, ks * 32, half);

    const f32x8 zero = {0.f, 0.f, 0.f, 0.f, 0.f, 0.f, 0.f, 0.f};
    f32x8 oacc[4] = {zero, zero, zero, zero};
    float mrun[8], lrun[8];
#pragma unroll
    for (int r = 0; r < 8; r++) { mrun[r] = -__builtin_inff(); lrun[r] = 0.f; }

    const int ldRow = tid >> 2;   // K: key row / V: dim row staged by thread
    const int ldSeg = tid & 3;    // 16-element segment of the 64-wide row

    // Stage K/V tile j into buffer buf: 4 async b128 copies per thread.
    auto stage_kv = [&](int j, int buf) {
        const bf16* kg = kb + (size_t)(j * 64 + ldRow) * HEAD_DIM + ldSeg * 16;
        bf16* kl = &kT[buf][ldRow * 64 + ldSeg * 16];
        cp_g2l_b128(kg, kl);
        cp_g2l_b128(kg + 8, kl + 8);
        const bf16* vg = vb + (size_t)ldRow * SEQ + j * 64 + ldSeg * 16;
        bf16* vl = &vT[buf][ldRow * 64 + ldSeg * 16];
        cp_g2l_b128(vg, vl);
        cp_g2l_b128(vg + 8, vl + 8);
    };

    const int NT = SEQ / 64;
    stage_kv(0, 0);

    for (int j = 0; j < NT; j++) {
        const int buf = j & 1;
        wait_async_all();          // tile j landed (per-wave ASYNCcnt)
        __syncthreads();           // all waves staged; prev readers of buf^1 done
        if (j + 1 < NT) stage_kv(j + 1, buf ^ 1);   // overlap with compute

        // S = Q K^T  (Q pre-scaled)  -- 8 WMMAs per wave
        f32x8 s[4] = {zero, zero, zero, zero};
#pragma unroll
        for (int ks = 0; ks < 2; ks++) {
#pragma unroll
            for (int nt = 0; nt < 4; nt++) {
                bf16x16 bfrag =
                    *(const bf16x16*)&kT[buf][(nt * 16 + l15) * 64 + ks * 32 + half * 16];
                s[nt] = wmma_bf16(qfrag[ks], bfrag, s[nt]);
            }
        }

        // Online softmax (rows spread over 16-lane halves; xor-shuffles stay in half)
#pragma unroll
        for (int r = 0; r < 8; r++) {
            float mx = fmaxf(fmaxf(s[0][r], s[1][r]), fmaxf(s[2][r], s[3][r]));
#pragma unroll
            for (int off = 1; off < 16; off <<= 1)
                mx = fmaxf(mx, __shfl_xor(mx, off, 32));
            const float mnew = fmaxf(mrun[r], mx);
            const float alpha = __expf(mrun[r] - mnew);
            mrun[r] = mnew;
            float rs = 0.f;
#pragma unroll
            for (int nt = 0; nt < 4; nt++) {
                float p = __expf(s[nt][r] - mnew);
                s[nt][r] = p;
                rs += p;
            }
#pragma unroll
            for (int off = 1; off < 16; off <<= 1)
                rs += __shfl_xor(rs, off, 32);
            lrun[r] = lrun[r] * alpha + rs;
#pragma unroll
            for (int dt = 0; dt < 4; dt++) oacc[dt][r] *= alpha;
        }

        // Re-layout P: C-tile layout -> A-fragment layout via per-wave LDS slab
#pragma unroll
        for (int nt = 0; nt < 4; nt++)
#pragma unroll
            for (int r = 0; r < 8; r++)
                pS[wave * 1024 + (r + 8 * half) * 64 + nt * 16 + l15] = (bf16)s[nt][r];
        __syncthreads();

        // O += P @ V  -- 8 WMMAs per wave, B-fragments contiguous from vT
#pragma unroll
        for (int ks = 0; ks < 2; ks++) {
            const bf16* pp = &pS[wave * 1024 + l15 * 64 + ks * 32 + half * 16];
            bf16x16 pf = combine8(*(const bf16x8*)pp, *(const bf16x8*)(pp + 8));
#pragma unroll
            for (int dt = 0; dt < 4; dt++) {
                bf16x16 vf =
                    *(const bf16x16*)&vT[buf][(dt * 16 + l15) * 64 + ks * 32 + half * 16];
                oacc[dt] = wmma_bf16(pf, vf, oacc[dt]);
            }
        }
    }

    // Normalize and store to [B*N, 512] bf16 (head-interleaved columns)
    const int b = bh >> 3, h = bh & 7;
#pragma unroll
    for (int r = 0; r < 8; r++) {
        const float inv = 1.0f / lrun[r];
        const int n = tile * 128 + wave * 16 + r + 8 * half;
        const size_t arow = (size_t)b * SEQ + n;
#pragma unroll
        for (int dt = 0; dt < 4; dt++)
            O[arow * DIM + h * HEAD_DIM + dt * 16 + l15] = (bf16)(oacc[dt][r] * inv);
    }
}

// ---------------------------------------------------------------------------
// Output projection: attn_out[8192,512]bf16 @ proj_w^T + proj_b -> f32 out
// Weight tile staged in LDS (two 64x256 halves) like qkv_gemm.
// ---------------------------------------------------------------------------
__global__ __launch_bounds__(256)
void proj_gemm_kernel(const bf16* __restrict__ A, const bf16* __restrict__ W,
                      const float* __restrict__ bias, float* __restrict__ out) {
    __shared__ bf16 wS[64 * 256];

    const int m0 = blockIdx.x * 128;
    const int n0 = blockIdx.y * 64;
    const int tid = threadIdx.x, wave = tid >> 5, lane = tid & 31;
    const int half = lane >> 4, l15 = lane & 15;
    const int mrow = m0 + wave * 16 + l15;

    const int lrow = tid >> 2;
    const int lseg = tid & 3;

    const f32x8 zero = {0.f, 0.f, 0.f, 0.f, 0.f, 0.f, 0.f, 0.f};
    f32x8 acc[4] = {zero, zero, zero, zero};

    for (int kk = 0; kk < DIM; kk += 256) {
#pragma unroll
        for (int u = 0; u < 8; u++)
            cp_g2l_b128(W + (size_t)(n0 + lrow) * DIM + kk + lseg * 64 + u * 8,
                        &wS[lrow * 256 + lseg * 64 + u * 8]);
        wait_async_all();
        __syncthreads();

        for (int k0 = 0; k0 < 256; k0 += 32) {
            bf16x16 afrag = load_a_frag(A + (size_t)mrow * DIM, kk + k0, half);
#pragma unroll
            for (int nt = 0; nt < 4; nt++) {
                bf16x16 bfrag =
                    *(const bf16x16*)&wS[(nt * 16 + l15) * 256 + k0 + half * 16];
                acc[nt] = wmma_bf16(afrag, bfrag, acc[nt]);
            }
        }
        __syncthreads();
    }

#pragma unroll
    for (int nt = 0; nt < 4; nt++) {
        const int col = n0 + nt * 16 + l15;
        const float bs = bias[col];
#pragma unroll
        for (int r = 0; r < 8; r++) {
            const int m = m0 + wave * 16 + r + 8 * half;
            out[(size_t)m * DIM + col] = acc[nt][r] + bs;
        }
    }
}

// ---------------------------------------------------------------------------
extern "C" void kernel_launch(void* const* d_in, const int* in_sizes, int n_in,
                              void* d_out, int out_size, void* d_ws, size_t ws_size,
                              hipStream_t stream) {
    const float* x      = (const float*)d_in[0];   // [2,4096,512]
    const float* qkv_w  = (const float*)d_in[1];   // [1536,512]
    const float* qkv_b  = (const float*)d_in[2];   // [1536]
    const float* proj_w = (const float*)d_in[3];   // [512,512]
    const float* proj_b = (const float*)d_in[4];   // [512]
    float* out = (float*)d_out;                    // [2,4096,512]

    const size_t ROWS  = (size_t)BATCH * SEQ;      // 8192
    const size_t XB    = ROWS * DIM;               // 4,194,304
    const size_t WQKV  = (size_t)3 * DIM * DIM;    //   786,432
    const size_t WPROJ = (size_t)DIM * DIM;        //   262,144
    const size_t QKVE  = ROWS * DIM;               // per-tensor Q/K/V elems

    char* w = (char*)d_ws;
    bf16* xb    = (bf16*)w;  w += XB    * sizeof(bf16);
    bf16* wqkvb = (bf16*)w;  w += WQKV  * sizeof(bf16);
    bf16* wprjb = (bf16*)w;  w += WPROJ * sizeof(bf16);
    bf16* qB    = (bf16*)w;  w += QKVE  * sizeof(bf16);
    bf16* kB    = (bf16*)w;  w += QKVE  * sizeof(bf16);
    bf16* vB    = (bf16*)w;  w += QKVE  * sizeof(bf16);   // transposed [B,H,Dh,N]
    bf16* aout  = (bf16*)w;  w += XB    * sizeof(bf16);

    f32_to_bf16_kernel<<<(XB / 8 + 255) / 256, 256, 0, stream>>>(x, xb, (int)XB);
    f32_to_bf16_kernel<<<(WQKV / 8 + 255) / 256, 256, 0, stream>>>(qkv_w, wqkvb, (int)WQKV);
    f32_to_bf16_kernel<<<(WPROJ / 8 + 255) / 256, 256, 0, stream>>>(proj_w, wprjb, (int)WPROJ);

    qkv_gemm_kernel<<<dim3(ROWS / 128, (3 * DIM) / 64), 256, 0, stream>>>(
        xb, wqkvb, qkv_b, qB, kB, vB);

    flash_attn_kernel<<<dim3(SEQ / 128, BATCH * HEADS), 256, 0, stream>>>(
        qB, kB, vB, aout);

    proj_gemm_kernel<<<dim3(ROWS / 128, DIM / 64), 256, 0, stream>>>(
        aout, wprjb, proj_b, out);
}
